// BiLSTM_AttentionModel_19662360281203
// MI455X (gfx1250) — matrix operand order repaired and checked by reading, not compiled, not run on hardware
//
#include <hip/hip_runtime.h>

// ---------------------------------------------------------------------------
// BiLSTM + MHA for MI455X (gfx1250, wave32, WMMA 16x16x32 bf16 + TDM staging)
// ---------------------------------------------------------------------------

typedef __attribute__((ext_vector_type(16))) __bf16       v16bf;
typedef __attribute__((ext_vector_type(8)))  __bf16       v8bf;
typedef __attribute__((ext_vector_type(8)))  float        v8f;
typedef __attribute__((ext_vector_type(4)))  unsigned int v4u;
typedef __attribute__((ext_vector_type(8)))  int          v8i;
typedef __attribute__((ext_vector_type(4)))  int          v4i;

#define B_   32
#define S_   512
#define I_   256
#define H_   512
#define E_   1024
#define NH_  16
#define HD_  64
#define G4H  (4 * H_)   // 2048

__device__ __forceinline__ v8f v8f_zero() {
  v8f z = {0.f, 0.f, 0.f, 0.f, 0.f, 0.f, 0.f, 0.f};
  return z;
}

__device__ __forceinline__ v16bf cat16(v8bf lo, v8bf hi) {
  return __builtin_shufflevector(lo, hi, 0, 1, 2, 3, 4, 5, 6, 7,
                                         8, 9, 10, 11, 12, 13, 14, 15);
}

// A fragment: 16 (rows) x 32 (K) bf16 tile from row-major src (leading dim ld).
// lane holds row M=lane&15, K range 16*(lane>>4) .. +15 (contiguous).
__device__ __forceinline__ v16bf load_a_frag(const __bf16* base, int ld,
                                             int kbase, int lane) {
  const int m   = lane & 15;
  const int sel = lane >> 4;
  const __bf16* p = base + (size_t)m * ld + kbase + 16 * sel;
  v8bf lo = *(const v8bf*)(p);
  v8bf hi = *(const v8bf*)(p + 8);
  return cat16(lo, hi);
}

// B fragment: B[k,n] = W[n,k], W row-major (N,K), base = &W[n0*ldw].
// lane holds column n = lane&15, K range 16*(lane>>4) .. +15 (contiguous).
__device__ __forceinline__ v16bf load_b_frag(const __bf16* base, int ldw,
                                             int kbase, int lane) {
  const int n   = lane & 15;
  const int sel = lane >> 4;
  const __bf16* p = base + (size_t)n * ldw + kbase + 16 * sel;
  v8bf lo = *(const v8bf*)(p);
  v8bf hi = *(const v8bf*)(p + 8);
  return cat16(lo, hi);
}

__device__ __forceinline__ v8f wmma_bf16(v16bf a, v16bf b, v8f c) {
  return __builtin_amdgcn_wmma_f32_16x16x32_bf16(
      /*neg_a=*/false, a, /*neg_b=*/false, b,
      /*c_mod=*/(short)0, c, /*reuse_a=*/false, /*reuse_b=*/false);
}

__device__ __forceinline__ float sigmoidf_(float x) {
  return 1.0f / (1.0f + __expf(-x));
}

// ---------------------------------------------------------------------------
// Tensor Data Mover: async 2D bf16 tile load Global -> LDS.
// D# per CDNA5 ISA:
//   group0: [1:0]count=1, [63:32]lds_addr, [120:64]global_addr, [127:126]type=2
//   group1: [17:16]data_size=1 (2B), [79:48]tensor_dim0, [111:80]tensor_dim1,
//           [127:112]tile_dim0, [143:128]tile_dim1, [207:160]dim0_stride
// Groups 2/3 zero (<=2D tensor).  Tracked with TENSORcnt.
// ---------------------------------------------------------------------------
__device__ __forceinline__ void tdm_load_2d_bf16(
    unsigned int lds_off, const void* gptr, unsigned int tensor_w,
    unsigned int tensor_h, unsigned int stride_elems, unsigned int tile_w,
    unsigned int tile_h) {
  const unsigned long long ga = (unsigned long long)(uintptr_t)gptr;
  v4u g0;
  g0[0] = 1u;                                        // count=1 (valid), user
  g0[1] = lds_off;                                   // lds_addr (bytes)
  g0[2] = (unsigned int)ga;                          // global_addr[31:0]
  g0[3] = (unsigned int)((ga >> 32) & 0x01FFFFFFu)   // global_addr[56:32]
          | (2u << 30);                              // type = 2 ("image")
  v8i g1;
  g1[0] = (int)(1u << 16);                           // data_size=1 (2 bytes)
  g1[1] = (int)((tensor_w & 0xFFFFu) << 16);         // tensor_dim0[15:0]
  g1[2] = (int)(((tensor_w >> 16) & 0xFFFFu) |       // tensor_dim0[31:16]
                ((tensor_h & 0xFFFFu) << 16));       // tensor_dim1[15:0]
  g1[3] = (int)(((tensor_h >> 16) & 0xFFFFu) |       // tensor_dim1[31:16]
                ((tile_w & 0xFFFFu) << 16));         // tile_dim0
  g1[4] = (int)(tile_h & 0xFFFFu);                   // tile_dim1 (tile_dim2=0)
  g1[5] = (int)stride_elems;                         // tensor_dim0_stride[31:0]
  g1[6] = 0;                                         // stride hi, dim1_stride lo
  g1[7] = 0;
  v4i z4 = {0, 0, 0, 0};
#if __clang_major__ >= 23
  v8i z8 = {0, 0, 0, 0, 0, 0, 0, 0};
  __builtin_amdgcn_tensor_load_to_lds(g0, g1, z4, z4, z8, 0);
#else
  __builtin_amdgcn_tensor_load_to_lds(g0, g1, z4, z4, 0);
#endif
}

// ---------------------------------------------------------------------------
// fp32 -> bf16 conversion
// ---------------------------------------------------------------------------
__global__ void cvt_f32_bf16(const float* __restrict__ src,
                             __bf16* __restrict__ dst, int n) {
  int i = blockIdx.x * 256 + threadIdx.x;
  if (i < n) dst[i] = (__bf16)src[i];
}

// ---------------------------------------------------------------------------
// WMMA GEMM:  C[M,N] = A[M,K](bf16) * W[N,K]^T(bf16) + bias[N]
// grid = (M/32, N/256); block = 256 (8 waves).
// A tile (32x32) staged via TDM into LDS; each wave computes a 32x32 block
// (2 M-tiles x 2 N-tiles) so B fragments are reused twice.
// ---------------------------------------------------------------------------
__global__ void __launch_bounds__(256)
wmma_gemm_bias(const __bf16* __restrict__ A, const __bf16* __restrict__ W,
               const float* __restrict__ bias, float* Cf, __bf16* Cb,
               int M, int N, int K) {
  __shared__ __bf16 aT[32 * 32];  // 2 KB
  const int lane = threadIdx.x & 31;
  const int wave = threadIdx.x >> 5;
  const int m0 = blockIdx.x * 32;
  const int n0 = blockIdx.y * 256 + wave * 32;
  v8f acc00 = v8f_zero(), acc01 = v8f_zero();
  v8f acc10 = v8f_zero(), acc11 = v8f_zero();
  for (int kb = 0; kb < K; kb += 32) {
    __syncthreads();  // previous tile's fragment reads complete
    if (wave == 0) {
      tdm_load_2d_bf16((unsigned int)(uintptr_t)aT,
                       A + (size_t)m0 * K + kb,
                       (unsigned int)K, (unsigned int)M, (unsigned int)K,
                       32u, 32u);
      __builtin_amdgcn_s_wait_tensorcnt((short)0);
    } else if (wave == 1 && lane == 0 && kb + 32 < K) {
      __builtin_prefetch(&W[(size_t)n0 * K + kb + 32], 0, 1);
    }
    __syncthreads();
    v16bf a0 = load_a_frag(aT, 32, 0, lane);
    v16bf a1 = load_a_frag(aT + 16 * 32, 32, 0, lane);
    v16bf b0 = load_b_frag(W + (size_t)n0 * K, K, kb, lane);
    v16bf b1 = load_b_frag(W + (size_t)(n0 + 16) * K, K, kb, lane);
    acc00 = wmma_bf16(a0, b0, acc00);
    acc01 = wmma_bf16(a0, b1, acc01);
    acc10 = wmma_bf16(a1, b0, acc10);
    acc11 = wmma_bf16(a1, b1, acc11);
  }
  const int sel = lane >> 4, nl = lane & 15;
#pragma unroll
  for (int r = 0; r < 8; ++r) {
    const int mA = m0 + r + 8 * sel;
    const int mB = mA + 16;
    const int c0 = n0 + nl, c1 = n0 + 16 + nl;
    const float v00 = acc00[r] + bias[c0];
    const float v01 = acc01[r] + bias[c1];
    const float v10 = acc10[r] + bias[c0];
    const float v11 = acc11[r] + bias[c1];
    if (Cf) {
      Cf[(size_t)mA * N + c0] = v00;
      Cf[(size_t)mA * N + c1] = v01;
      Cf[(size_t)mB * N + c0] = v10;
      Cf[(size_t)mB * N + c1] = v11;
    }
    if (Cb) {
      Cb[(size_t)mA * N + c0] = (__bf16)v00;
      Cb[(size_t)mA * N + c1] = (__bf16)v01;
      Cb[(size_t)mB * N + c0] = (__bf16)v10;
      Cb[(size_t)mB * N + c1] = (__bf16)v11;
    }
  }
}

// ---------------------------------------------------------------------------
// BiLSTM recurrence. grid = 2 (dir), block = 1024 (32 waves).
// Per step: gates(32x2048) = xg[:,t,:] + h(32x512) @ Whh^T via WMMA.
// h kept bf16 in LDS; cell state c kept in accumulator fragments all 512 steps.
// ---------------------------------------------------------------------------
__global__ void __launch_bounds__(1024)
bilstm_recurrence(const float* __restrict__ xg_f, const float* __restrict__ xg_b,
                  const __bf16* __restrict__ Whh_f,
                  const __bf16* __restrict__ Whh_b,
                  __bf16* __restrict__ h_all) {
  __shared__ __bf16 hbuf[B_ * H_];  // 32 KB
  const int dir = blockIdx.x;
  const float* xg = dir ? xg_b : xg_f;
  const __bf16* Whh = dir ? Whh_b : Whh_f;
  const int lane = threadIdx.x & 31;
  const int wave = threadIdx.x >> 5;
  const int mt = wave & 1;
  const int jt0 = wave >> 1;
  const int sel = lane >> 4, nl = lane & 15;

  for (int i = threadIdx.x; i < B_ * H_; i += 1024) hbuf[i] = (__bf16)0.0f;
  v8f cfrag[2] = {v8f_zero(), v8f_zero()};
  __syncthreads();

  for (int step = 0; step < S_; ++step) {
    const int t = dir ? (S_ - 1 - step) : step;
    v8f hout[2];
#pragma unroll
    for (int p = 0; p < 2; ++p) {
      const int jt = jt0 + 16 * p;
      v8f acc[4] = {v8f_zero(), v8f_zero(), v8f_zero(), v8f_zero()};
      for (int kb = 0; kb < H_; kb += 32) {
        v16bf af = load_a_frag(hbuf + mt * 16 * H_, H_, kb, lane);
#pragma unroll
        for (int g = 0; g < 4; ++g) {
          v16bf bf = load_b_frag(Whh + (size_t)(g * H_ + jt * 16) * H_, H_,
                                 kb, lane);
          acc[g] = wmma_bf16(af, bf, acc[g]);
        }
      }
#pragma unroll
      for (int r = 0; r < 8; ++r) {
        const int b = mt * 16 + r + 8 * sel;
        const size_t xrow = ((size_t)b * S_ + t) * G4H + jt * 16 + nl;
        const float gi = acc[0][r] + xg[xrow + 0 * H_];
        const float gf = acc[1][r] + xg[xrow + 1 * H_];
        const float gg = acc[2][r] + xg[xrow + 2 * H_];
        const float go = acc[3][r] + xg[xrow + 3 * H_];
        const float c = sigmoidf_(gf) * cfrag[p][r] + sigmoidf_(gi) * tanhf(gg);
        cfrag[p][r] = c;
        hout[p][r] = sigmoidf_(go) * tanhf(c);
      }
    }
    __syncthreads();  // all waves done reading hbuf for step t
#pragma unroll
    for (int p = 0; p < 2; ++p) {
      const int jt = jt0 + 16 * p;
#pragma unroll
      for (int r = 0; r < 8; ++r) {
        const int b = mt * 16 + r + 8 * sel;
        const int j = jt * 16 + nl;
        const __bf16 hv = (__bf16)hout[p][r];
        hbuf[b * H_ + j] = hv;
        h_all[((size_t)b * S_ + t) * E_ + dir * H_ + j] = hv;
      }
    }
    __syncthreads();  // h for step t visible before next step's reads
  }
}

// ---------------------------------------------------------------------------
// Multi-head attention. grid = B*NH*(S/16) = 16384 blocks; block = 128.
// TDM prefetches the (512 x 64) V head-slice into LDS at kernel start and the
// wait is deferred until after softmax (async copy/compute overlap).
// Phase1: scores 16x512 via WMMA into LDS.  Phase2: softmax.
// Phase3: out 16x64 = attn @ V(LDS) via WMMA.
// ---------------------------------------------------------------------------
__global__ void __launch_bounds__(128)
mha_kernel(const __bf16* __restrict__ Q, const __bf16* __restrict__ Km,
           const __bf16* __restrict__ V, __bf16* __restrict__ O) {
  __shared__ float  sc[16 * S_];     // 32 KB scores
  __shared__ __bf16 vbuf[S_ * HD_];  // 64 KB V slice
  const int qt = blockIdx.x & 31;
  const int head = (blockIdx.x >> 5) & 15;
  const int b = blockIdx.x >> 9;
  const int lane = threadIdx.x & 31;
  const int wave = threadIdx.x >> 5;
  const int sel = lane >> 4, nl = lane & 15;
  const size_t hoff = (size_t)head * HD_;
  const __bf16* qbase = Q + ((size_t)b * S_ + qt * 16) * E_ + hoff;

  // Kick off async V-slice load (tile 64 wide x 512 rows, row stride E).
  if (wave == 0) {
    tdm_load_2d_bf16((unsigned int)(uintptr_t)vbuf,
                     V + (size_t)b * S_ * E_ + hoff,
                     (unsigned int)E_, (unsigned int)S_, (unsigned int)E_,
                     (unsigned int)HD_, (unsigned int)S_);
  }

  // Phase 1: scores = Q K^T / sqrt(HD).  Each wave does 8 k-tiles (128 cols).
  for (int kt = wave * 8; kt < wave * 8 + 8; ++kt) {
    v8f acc = v8f_zero();
    const __bf16* kbase = Km + ((size_t)b * S_ + kt * 16) * E_ + hoff;
#pragma unroll
    for (int d = 0; d < HD_; d += 32) {
      v16bf af = load_a_frag(qbase, E_, d, lane);
      v16bf bf = load_b_frag(kbase, E_, d, lane);
      acc = wmma_bf16(af, bf, acc);
    }
#pragma unroll
    for (int r = 0; r < 8; ++r)
      sc[(r + 8 * sel) * S_ + kt * 16 + nl] = acc[r] * 0.125f;
  }
  __syncthreads();

  // Phase 2: row softmax over 512 keys.
  if (threadIdx.x < 16) {
    float* row = &sc[threadIdx.x * S_];
    float mx = row[0];
    for (int k = 1; k < S_; ++k) mx = fmaxf(mx, row[k]);
    float sum = 0.f;
    for (int k = 0; k < S_; ++k) {
      const float e = __expf(row[k] - mx);
      row[k] = e;
      sum += e;
    }
    const float inv = 1.0f / sum;
    for (int k = 0; k < S_; ++k) row[k] *= inv;
  }
  if (wave == 0) __builtin_amdgcn_s_wait_tensorcnt((short)0);
  __syncthreads();  // V slice resident + softmax published

  // Phase 3: out = attn @ V.  Each wave owns one 16-wide slice of HD.
  const int n0d = wave * 16;
  v8f acc = v8f_zero();
  for (int kb = 0; kb < S_; kb += 32) {
    v16bf af;
#pragma unroll
    for (int j = 0; j < 16; ++j) {
      const int kk = kb + 16 * sel + j;
      af[j] = (__bf16)sc[(lane & 15) * S_ + kk];
    }
    v16bf bf;
#pragma unroll
    for (int j = 0; j < 16; ++j) {
      const int s = kb + 16 * sel + j;
      bf[j] = vbuf[s * HD_ + n0d + nl];
    }
    acc = wmma_bf16(af, bf, acc);
  }
#pragma unroll
  for (int r = 0; r < 8; ++r) {
    const int q = qt * 16 + r + 8 * sel;
    O[((size_t)b * S_ + q) * E_ + hoff + n0d + nl] = (__bf16)acc[r];
  }
}

// ---------------------------------------------------------------------------
// Final: only the last timestep matters.
// out[b] = sigmoid( sum_n (attn_row . Wo[n,:] + bo[n]) * Wfc[n] + bfc )
// ---------------------------------------------------------------------------
__global__ void __launch_bounds__(256)
final_proj_fc(const __bf16* __restrict__ O, const float* __restrict__ Wo,
              const float* __restrict__ bo, const float* __restrict__ Wfc,
              const float* __restrict__ bfc, float* __restrict__ out) {
  __shared__ float red[256];
  const int b = blockIdx.x;
  const __bf16* row = O + ((size_t)b * S_ + (S_ - 1)) * E_;
  float partial = 0.f;
  for (int n = threadIdx.x; n < E_; n += 256) {
    float dot = bo[n];
    const float* w = Wo + (size_t)n * E_;
    for (int k = 0; k < E_; ++k) dot += (float)row[k] * w[k];
    partial += dot * Wfc[n];
  }
  red[threadIdx.x] = partial;
  __syncthreads();
  for (int s = 128; s > 0; s >>= 1) {
    if (threadIdx.x < s) red[threadIdx.x] += red[threadIdx.x + s];
    __syncthreads();
  }
  if (threadIdx.x == 0) out[b] = sigmoidf_(red[0] + bfc[0]);
}

// ---------------------------------------------------------------------------
// Host launcher
// ---------------------------------------------------------------------------
extern "C" void kernel_launch(void* const* d_in, const int* in_sizes, int n_in,
                              void* d_out, int out_size, void* d_ws,
                              size_t ws_size, hipStream_t stream) {
  (void)in_sizes; (void)n_in; (void)out_size; (void)ws_size;
  const float* x     = (const float*)d_in[0];
  const float* Wih_f = (const float*)d_in[1];
  const float* Whh_f = (const float*)d_in[2];
  const float* b_f   = (const float*)d_in[3];
  const float* Wih_b = (const float*)d_in[4];
  const float* Whh_b = (const float*)d_in[5];
  const float* b_b   = (const float*)d_in[6];
  const float* Wq    = (const float*)d_in[7];
  const float* bq    = (const float*)d_in[8];
  const float* Wk    = (const float*)d_in[9];
  const float* bk    = (const float*)d_in[10];
  const float* Wv    = (const float*)d_in[11];
  const float* bv    = (const float*)d_in[12];
  const float* Wo    = (const float*)d_in[13];
  const float* bo    = (const float*)d_in[14];
  const float* Wfc   = (const float*)d_in[15];
  const float* bfc   = (const float*)d_in[16];
  float* out = (float*)d_out;

  char* ws = (char*)d_ws;
  size_t off = 0;
  auto alloc = [&](size_t bytes) -> char* {
    char* p = ws + off;
    off += (bytes + 255) & ~(size_t)255;
    return p;
  };

  const size_t nX   = (size_t)B_ * S_ * I_;
  const size_t nWih = (size_t)G4H * I_;
  const size_t nWhh = (size_t)G4H * H_;
  const size_t nWe  = (size_t)E_ * E_;
  const size_t nAct = (size_t)B_ * S_ * E_;
  const size_t nXg  = (size_t)B_ * S_ * G4H;

  __bf16* x_bf     = (__bf16*)alloc(nX * 2);
  __bf16* Wih_f_bf = (__bf16*)alloc(nWih * 2);
  __bf16* Whh_f_bf = (__bf16*)alloc(nWhh * 2);
  __bf16* Wih_b_bf = (__bf16*)alloc(nWih * 2);
  __bf16* Whh_b_bf = (__bf16*)alloc(nWhh * 2);
  __bf16* Wq_bf    = (__bf16*)alloc(nWe * 2);
  __bf16* Wk_bf    = (__bf16*)alloc(nWe * 2);
  __bf16* Wv_bf    = (__bf16*)alloc(nWe * 2);
  float*  xg_f     = (float*)alloc(nXg * 4);
  float*  xg_b     = (float*)alloc(nXg * 4);
  __bf16* h_bf     = (__bf16*)alloc(nAct * 2);
  __bf16* Q_bf     = (__bf16*)alloc(nAct * 2);
  __bf16* K_bf     = (__bf16*)alloc(nAct * 2);
  __bf16* V_bf     = (__bf16*)alloc(nAct * 2);
  __bf16* O_bf     = (__bf16*)alloc(nAct * 2);

  auto cvt = [&](const float* s, __bf16* d, size_t n) {
    cvt_f32_bf16<<<dim3((unsigned)((n + 255) / 256)), dim3(256), 0, stream>>>(
        s, d, (int)n);
  };
  cvt(x, x_bf, nX);
  cvt(Wih_f, Wih_f_bf, nWih);
  cvt(Whh_f, Whh_f_bf, nWhh);
  cvt(Wih_b, Wih_b_bf, nWih);
  cvt(Whh_b, Whh_b_bf, nWhh);
  cvt(Wq, Wq_bf, nWe);
  cvt(Wk, Wk_bf, nWe);
  cvt(Wv, Wv_bf, nWe);

  const int M = B_ * S_;  // 16384
  // xg = x @ W_ih^T + b  (fp32 out, consumed by recurrence)
  wmma_gemm_bias<<<dim3(M / 32, G4H / 256), dim3(256), 0, stream>>>(
      x_bf, Wih_f_bf, b_f, xg_f, (__bf16*)nullptr, M, G4H, I_);
  wmma_gemm_bias<<<dim3(M / 32, G4H / 256), dim3(256), 0, stream>>>(
      x_bf, Wih_b_bf, b_b, xg_b, (__bf16*)nullptr, M, G4H, I_);

  // Sequential bidirectional recurrence (one persistent block per direction).
  bilstm_recurrence<<<dim3(2), dim3(1024), 0, stream>>>(
      xg_f, xg_b, Whh_f_bf, Whh_b_bf, h_bf);

  // Q/K/V projections (bf16 out, consumed by attention).
  wmma_gemm_bias<<<dim3(M / 32, E_ / 256), dim3(256), 0, stream>>>(
      h_bf, Wq_bf, bq, (float*)nullptr, Q_bf, M, E_, E_);
  wmma_gemm_bias<<<dim3(M / 32, E_ / 256), dim3(256), 0, stream>>>(
      h_bf, Wk_bf, bk, (float*)nullptr, K_bf, M, E_, E_);
  wmma_gemm_bias<<<dim3(M / 32, E_ / 256), dim3(256), 0, stream>>>(
      h_bf, Wv_bf, bv, (float*)nullptr, V_bf, M, E_, E_);

  // Attention.
  mha_kernel<<<dim3(B_ * NH_ * (S_ / 16)), dim3(128), 0, stream>>>(
      Q_bf, K_bf, V_bf, O_bf);

  // Output projection (last timestep only) + FC + sigmoid.
  final_proj_fc<<<dim3(B_), dim3(256), 0, stream>>>(O_bf, Wo, bo, Wfc, bfc,
                                                    out);
}
